// TransformerBlockQuantum_65481071399629
// MI455X (gfx1250) — matrix operand order repaired and checked by reading, not compiled, not run on hardware
//
#include <hip/hip_runtime.h>
#include <hip/hip_bf16.h>

typedef __attribute__((ext_vector_type(16))) __bf16 v16bf;
typedef __attribute__((ext_vector_type(8)))  float  v8f;

#define B_DIM 2
#define S_DIM 1024
#define E_DIM 512
#define H_DIM 64
#define F_DIM 2048
#define ROWS  (B_DIM * S_DIM)   // 2048

// GEMM tiling
#define BM 128
#define BN 64
#define BK 64
#define LDA 66   // padded LDS stride (elements) for A [row][k]
#define LDB 66   // padded LDS stride (elements) for Bt [n][k]

// ---------- helpers ----------
__device__ __forceinline__ unsigned short f2bf(float f) {
  union { float f; unsigned u; } v; v.f = f;
  unsigned u = v.u;
  u += 0x7FFFu + ((u >> 16) & 1u);      // round-to-nearest-even
  return (unsigned short)(u >> 16);
}

// async copy of 16 bytes global -> LDS (no VGPR data round-trip), ASYNCcnt-tracked
__device__ __forceinline__ void async_g2l_b128(void* lds_ptr, const void* gptr) {
  unsigned lds = (unsigned)(size_t)lds_ptr;                 // LDS address from low 32 bits
  unsigned long long ga = (unsigned long long)(size_t)gptr;
  asm volatile("global_load_async_to_lds_b128 %0, %1, off"
               :: "v"(lds), "v"(ga) : "memory");
}
__device__ __forceinline__ void wait_async0() {
  asm volatile("s_wait_asynccnt 0" ::: "memory");
}

// ---------- weight f32 -> bf16 ----------
__global__ __launch_bounds__(256) void cvt_bf16_kernel(
    const float* __restrict__ in, unsigned short* __restrict__ out, int n) {
  int i = blockIdx.x * 256 + threadIdx.x;
  if (i < n) out[i] = f2bf(in[i]);
}

// ---------- quantum attention circuit: ring expvals ----------
__global__ __launch_bounds__(256) void qkv_kernel(
    const float* __restrict__ x, const float* __restrict__ theta,
    float* __restrict__ q) {
  int idx = blockIdx.x * 256 + threadIdx.x;          // over B*S*H
  int h  = idx & (H_DIM - 1);
  int bs = idx >> 6;
  const float* xp = x + (size_t)bs * E_DIM + h * 8;
  float c[8];
#pragma unroll
  for (int d = 0; d < 8; ++d) c[d] = __cosf(xp[d] + theta[d]);
  float out[8];
  float p1 = 1.f;
#pragma unroll
  for (int d = 1; d < 8; ++d) p1 *= c[d];
  out[0] = p1;                                       // prod_{j>=1}
  float run = c[0];
#pragma unroll
  for (int d = 1; d < 8; ++d) { run *= c[d]; out[d] = run; }  // cumprod
  int b = bs >> 10, s = bs & (S_DIM - 1);
  float* qp = q + (((size_t)(b * H_DIM + h)) * S_DIM + s) * 8;
#pragma unroll
  for (int d = 0; d < 8; ++d) qp[d] = out[d];
}

// ---------- flash-style attention per (b,h) ----------
__global__ __launch_bounds__(256) void attn_kernel(
    const float* __restrict__ q, unsigned short* __restrict__ arows) {
  __shared__ float4 sq[S_DIM * 2];                   // [t][8] f32 = 32 KB
  int bh = blockIdx.x;                               // b*H + h
  int b = bh >> 6, h = bh & 63;
  const float4* gq = (const float4*)(q + (size_t)bh * S_DIM * 8);
  for (int i = threadIdx.x; i < S_DIM * 2; i += 256) sq[i] = gq[i];
  __syncthreads();
  const float scale = 0.35355339059327379f;          // 1/sqrt(8)
  for (int rr = 0; rr < 4; ++rr) {
    int s = rr * 256 + threadIdx.x;
    float4 qa = sq[s * 2], qb = sq[s * 2 + 1];
    float m = -3.0e38f, l = 0.f;
    float acc[8] = {};
    for (int t = 0; t < S_DIM; ++t) {                // LDS broadcast reads
      float4 ka = sq[t * 2], kb = sq[t * 2 + 1];
      float dot = qa.x*ka.x + qa.y*ka.y + qa.z*ka.z + qa.w*ka.w
                + qb.x*kb.x + qb.y*kb.y + qb.z*kb.z + qb.w*kb.w;
      float sc   = dot * scale;
      float mn   = fmaxf(m, sc);
      float corr = __expf(m - mn);
      float p    = __expf(sc - mn);
      l = l * corr + p;
      acc[0] = acc[0]*corr + p*ka.x;  acc[1] = acc[1]*corr + p*ka.y;
      acc[2] = acc[2]*corr + p*ka.z;  acc[3] = acc[3]*corr + p*ka.w;
      acc[4] = acc[4]*corr + p*kb.x;  acc[5] = acc[5]*corr + p*kb.y;
      acc[6] = acc[6]*corr + p*kb.z;  acc[7] = acc[7]*corr + p*kb.w;
      m = mn;
    }
    float inv = 1.f / l;
    unsigned short* op = arows + ((size_t)(b * S_DIM + s)) * E_DIM + h * 8;
#pragma unroll
    for (int d = 0; d < 8; ++d) op[d] = f2bf(acc[d] * inv);
  }
}

// ---------- A fragment (16x32 bf16) from LDS [row][k] ----------
// lane half kh holds K in {k0+16kh..+15}, contiguous
__device__ __forceinline__ v16bf load_fragA(const unsigned short* sA,
                                            int row0, int r, int kh, int k0) {
  union { unsigned u[8]; v16bf v; } f;
  const unsigned short* ap = sA + (row0 + r) * LDA + k0;
  const unsigned* p0 = (const unsigned*)(ap + 16 * kh);
  const unsigned* p1 = (const unsigned*)(ap + 16 * kh + 8);
#pragma unroll
  for (int d = 0; d < 4; ++d) { f.u[d] = p0[d]; f.u[4 + d] = p1[d]; }
  return f.v;
}

// ---------- B fragment (32x16 bf16) from transposed LDS [n][k] ----------
// lane half kh holds K in {k0+16kh..+15}, contiguous
__device__ __forceinline__ v16bf load_fragB(const unsigned short* sBt,
                                            int n0, int r, int kh, int k0) {
  union { unsigned u[8]; v16bf v; } f;
  const unsigned* p = (const unsigned*)(sBt + (n0 + r) * LDB + k0 + 16 * kh);
#pragma unroll
  for (int d = 0; d < 8; ++d) f.u[d] = p[d];
  return f.v;
}

// ---------- bf16 WMMA GEMM: C[M,N] = A[M,K] * B[K,N] ----------
// Block tile 128x64, BK=64; 8 waves in 4x2 grid; each wave computes a 32x32
// register tile as 2x2 v_wmma_f32_16x16x32_bf16 (A/B fragment reuse).
// A tile staged with GLOBAL_LOAD_ASYNC_TO_LDS_B128 (ASYNCcnt); B staged
// transposed via VGPRs. EPI 0: store f32. EPI 1: +bias, ReLU, store bf16.
template <int EPI>
__global__ __launch_bounds__(256) void gemm_bf16_kernel(
    const unsigned short* __restrict__ A, const unsigned short* __restrict__ Bw,
    const float* __restrict__ bias, float* __restrict__ Cf,
    unsigned short* __restrict__ Cb, int M, int N, int K) {
  __shared__ unsigned short sA[BM * LDA];             // [row][k]
  __shared__ unsigned short sBt[BN * LDB];            // [n][k]
  const int tid  = threadIdx.x;
  const int lane = tid & 31;
  const int wave = tid >> 5;
  const int wm   = wave & 3;                          // 32-row strip
  const int wn   = wave >> 2;                          // 32-col strip
  const int rowBase = blockIdx.y * BM;
  const int colBase = blockIdx.x * BN;

  const int r  = lane & 15;
  const int kh = lane >> 4;

  v8f c00 = {}, c01 = {}, c10 = {}, c11 = {};

  for (int kb = 0; kb < K; kb += BK) {
    if (kb + BK < K) {                                // -> global_prefetch_b8
      __builtin_prefetch(Bw + (size_t)(kb + BK + (tid >> 3)) * N + colBase + (tid & 7) * 8, 0, 3);
    }
    // stage A tile 128x64 (row-major): async DMA global -> LDS, no VGPR data
#pragma unroll
    for (int i = 0; i < 4; ++i) {
      int idx = tid + i * 256;
      int row = idx >> 3, chunk = idx & 7;
      async_g2l_b128(sA + row * LDA + chunk * 8,
                     A + (size_t)(rowBase + row) * K + kb + chunk * 8);
    }
    // stage B tile 64x64 transposed -> [n][k], 2 uint4 per thread
#pragma unroll
    for (int i = 0; i < 2; ++i) {
      int idx = tid + i * 256;
      int krow = idx >> 3, chunk = idx & 7;
      uint4 bv = *(const uint4*)(Bw + (size_t)(kb + krow) * N + colBase + chunk * 8);
      unsigned bw_[4] = { bv.x, bv.y, bv.z, bv.w };
#pragma unroll
      for (int j = 0; j < 4; ++j) {
        sBt[(chunk * 8 + 2 * j    ) * LDB + krow] = (unsigned short)(bw_[j] & 0xFFFFu);
        sBt[(chunk * 8 + 2 * j + 1) * LDB + krow] = (unsigned short)(bw_[j] >> 16);
      }
    }
    wait_async0();                                    // A-tile DMA complete
    __syncthreads();

#pragma unroll
    for (int ks = 0; ks < BK; ks += 32) {
      v16bf fa0 = load_fragA(sA, wm * 32,      r, kh, ks);
      v16bf fa1 = load_fragA(sA, wm * 32 + 16, r, kh, ks);
      v16bf fb0 = load_fragB(sBt, wn * 32,      r, kh, ks);
      v16bf fb1 = load_fragB(sBt, wn * 32 + 16, r, kh, ks);
      c00 = __builtin_amdgcn_wmma_f32_16x16x32_bf16(false, fa0, false, fb0, (short)0, c00, false, false);
      c01 = __builtin_amdgcn_wmma_f32_16x16x32_bf16(false, fa0, false, fb1, (short)0, c01, false, false);
      c10 = __builtin_amdgcn_wmma_f32_16x16x32_bf16(false, fa1, false, fb0, (short)0, c10, false, false);
      c11 = __builtin_amdgcn_wmma_f32_16x16x32_bf16(false, fa1, false, fb1, (short)0, c11, false, false);
    }
    __syncthreads();
  }

  // C layout: element i of v8f -> M = i + 8*kh, N = lane&15
  const int gr0 = rowBase + wm * 32 + 8 * kh;
  const int gr1 = gr0 + 16;
  const int gc0 = colBase + wn * 32 + r;
  const int gc1 = gc0 + 16;
#pragma unroll
  for (int i = 0; i < 8; ++i) {
    size_t o00 = (size_t)(gr0 + i) * N + gc0;
    size_t o01 = (size_t)(gr0 + i) * N + gc1;
    size_t o10 = (size_t)(gr1 + i) * N + gc0;
    size_t o11 = (size_t)(gr1 + i) * N + gc1;
    if (EPI == 1) {
      float b0 = bias[gc0], b1 = bias[gc1];
      Cb[o00] = f2bf(fmaxf(c00[i] + b0, 0.f));
      Cb[o01] = f2bf(fmaxf(c01[i] + b1, 0.f));
      Cb[o10] = f2bf(fmaxf(c10[i] + b0, 0.f));
      Cb[o11] = f2bf(fmaxf(c11[i] + b1, 0.f));
    } else {
      Cf[o00] = c00[i]; Cf[o01] = c01[i];
      Cf[o10] = c10[i]; Cf[o11] = c11[i];
    }
  }
}

// ---------- LN1: x1 = LN(x + (Y + b_combine)); qf = bf16(cos(x1)*cos(theta)) ----------
__global__ __launch_bounds__(256) void ln1_kernel(
    const float* __restrict__ x, const float* __restrict__ y,
    const float* __restrict__ bc, const float* __restrict__ lnw,
    const float* __restrict__ lnb, const float* __restrict__ theta,
    float* __restrict__ x1, unsigned short* __restrict__ qf) {
  __shared__ float s1[256], s2[256];
  int row = blockIdx.x, t = threadIdx.x;
  size_t base = (size_t)row * E_DIM;
  float v0 = x[base + t]       + y[base + t]       + bc[t];
  float v1 = x[base + t + 256] + y[base + t + 256] + bc[t + 256];
  s1[t] = v0 + v1;
  s2[t] = v0 * v0 + v1 * v1;
  __syncthreads();
  for (int o = 128; o > 0; o >>= 1) {
    if (t < o) { s1[t] += s1[t + o]; s2[t] += s2[t + o]; }
    __syncthreads();
  }
  float mu  = s1[0] * (1.f / E_DIM);
  float var = s2[0] * (1.f / E_DIM) - mu * mu;
  float rs  = rsqrtf(var + 1e-5f);
  float o0 = (v0 - mu) * rs * lnw[t]       + lnb[t];
  float o1 = (v1 - mu) * rs * lnw[t + 256] + lnb[t + 256];
  x1[base + t]       = o0;
  x1[base + t + 256] = o1;
  qf[base + t]       = f2bf(__cosf(o0) * __cosf(theta[t]));
  qf[base + t + 256] = f2bf(__cosf(o1) * __cosf(theta[t + 256]));
}

// ---------- LN2: out = LN(x1 + (Y2 + b2)) ----------
__global__ __launch_bounds__(256) void ln2_kernel(
    const float* __restrict__ x1, const float* __restrict__ y2,
    const float* __restrict__ b2, const float* __restrict__ lnw,
    const float* __restrict__ lnb, float* __restrict__ out) {
  __shared__ float s1[256], s2[256];
  int row = blockIdx.x, t = threadIdx.x;
  size_t base = (size_t)row * E_DIM;
  float v0 = x1[base + t]       + y2[base + t]       + b2[t];
  float v1 = x1[base + t + 256] + y2[base + t + 256] + b2[t + 256];
  s1[t] = v0 + v1;
  s2[t] = v0 * v0 + v1 * v1;
  __syncthreads();
  for (int o = 128; o > 0; o >>= 1) {
    if (t < o) { s1[t] += s1[t + o]; s2[t] += s2[t + o]; }
    __syncthreads();
  }
  float mu  = s1[0] * (1.f / E_DIM);
  float var = s2[0] * (1.f / E_DIM) - mu * mu;
  float rs  = rsqrtf(var + 1e-5f);
  out[base + t]       = (v0 - mu) * rs * lnw[t]       + lnb[t];
  out[base + t + 256] = (v1 - mu) * rs * lnw[t + 256] + lnb[t + 256];
}

// ---------- host ----------
extern "C" void kernel_launch(void* const* d_in, const int* in_sizes, int n_in,
                              void* d_out, int out_size, void* d_ws, size_t ws_size,
                              hipStream_t stream) {
  const float* x       = (const float*)d_in[0];
  const float* th_attn = (const float*)d_in[1];
  const float* w_comb  = (const float*)d_in[2];
  const float* b_comb  = (const float*)d_in[3];
  const float* th_ffn  = (const float*)d_in[4];
  const float* w1      = (const float*)d_in[5];
  const float* b1      = (const float*)d_in[6];
  const float* w2      = (const float*)d_in[7];
  const float* b2      = (const float*)d_in[8];
  const float* ln1w    = (const float*)d_in[9];
  const float* ln1b    = (const float*)d_in[10];
  const float* ln2w    = (const float*)d_in[11];
  const float* ln2b    = (const float*)d_in[12];
  float* out = (float*)d_out;

  char* ws = (char*)d_ws;
  size_t off = 0;
  auto alloc = [&](size_t bytes) -> void* {
    void* p = ws + off;
    off += (bytes + 255) & ~(size_t)255;
    return p;
  };
  float*          q_f32 = (float*)alloc((size_t)B_DIM * H_DIM * S_DIM * 8 * 4); // 4 MB
  unsigned short* arows = (unsigned short*)alloc((size_t)ROWS * E_DIM * 2);     // 2 MB
  unsigned short* wcb   = (unsigned short*)alloc((size_t)E_DIM * E_DIM * 2);
  unsigned short* w1b   = (unsigned short*)alloc((size_t)E_DIM * F_DIM * 2);
  unsigned short* w2b   = (unsigned short*)alloc((size_t)F_DIM * E_DIM * 2);
  float*          y1    = (float*)alloc((size_t)ROWS * E_DIM * 4);
  float*          x1    = (float*)alloc((size_t)ROWS * E_DIM * 4);
  unsigned short* qf    = (unsigned short*)alloc((size_t)ROWS * E_DIM * 2);
  unsigned short* hdn   = (unsigned short*)alloc((size_t)ROWS * F_DIM * 2);     // 8 MB
  float*          y2    = (float*)alloc((size_t)ROWS * E_DIM * 4);

  // weight conversion
  cvt_bf16_kernel<<<(E_DIM * E_DIM) / 256, 256, 0, stream>>>(w_comb, wcb, E_DIM * E_DIM);
  cvt_bf16_kernel<<<(E_DIM * F_DIM) / 256, 256, 0, stream>>>(w1, w1b, E_DIM * F_DIM);
  cvt_bf16_kernel<<<(F_DIM * E_DIM) / 256, 256, 0, stream>>>(w2, w2b, F_DIM * E_DIM);

  // quantum circuit expvals
  qkv_kernel<<<(B_DIM * S_DIM * H_DIM) / 256, 256, 0, stream>>>(x, th_attn, q_f32);

  // attention (flash-style, per head)
  attn_kernel<<<B_DIM * H_DIM, 256, 0, stream>>>(q_f32, arows);

  // combine GEMM: [2048,512] x [512,512] -> f32
  gemm_bf16_kernel<0><<<dim3(E_DIM / BN, ROWS / BM), 256, 0, stream>>>(
      arows, wcb, nullptr, y1, nullptr, ROWS, E_DIM, E_DIM);

  // LN1 + quantum FFN input
  ln1_kernel<<<ROWS, 256, 0, stream>>>(x, y1, b_comb, ln1w, ln1b, th_ffn, x1, qf);

  // FFN1: [2048,2048] = relu(qf x w1 + b1) -> bf16
  gemm_bf16_kernel<1><<<dim3(F_DIM / BN, ROWS / BM), 256, 0, stream>>>(
      qf, w1b, b1, nullptr, hdn, ROWS, F_DIM, E_DIM);

  // FFN2: [2048,512] = hdn x w2 -> f32
  gemm_bf16_kernel<0><<<dim3(E_DIM / BN, ROWS / BM), 256, 0, stream>>>(
      hdn, w2b, nullptr, y2, nullptr, ROWS, E_DIM, F_DIM);

  // LN2 -> output
  ln2_kernel<<<ROWS, 256, 0, stream>>>(x1, y2, b2, ln2w, ln2b, out);
}
